// MultiHeadAttention_12859132084765
// MI455X (gfx1250) — matrix operand order repaired and checked by reading, not compiled, not run on hardware
//
#include <hip/hip_runtime.h>
#include <hip/hip_bf16.h>

typedef __attribute__((ext_vector_type(16))) __bf16 bf16x16;
typedef __attribute__((ext_vector_type(8)))  __bf16 bf16x8;
typedef __attribute__((ext_vector_type(4)))  __bf16 bf16x4;
typedef __attribute__((ext_vector_type(8)))  float  f32x8;
typedef __attribute__((ext_vector_type(4)))  float  f32x4;
typedef __attribute__((ext_vector_type(4)))  int    i32x4;

#define EMB   1024
#define SEQ   1024
#define BATCH 8
#define HEADS 8
#define HD    128

__device__ __forceinline__ f32x8 wmma_bf16(bf16x16 a, bf16x16 b, f32x8 c) {
    // D = A(16x32 bf16) * B(32x16 bf16) + C(16x16 f32)
    return __builtin_amdgcn_wmma_f32_16x16x32_bf16(false, a, false, b,
                                                   (short)0, c, false, false);
}

__device__ __forceinline__ bf16x16 cat8(bf16x8 lo, bf16x8 hi) {
    return __builtin_shufflevector(lo, hi, 0,1,2,3,4,5,6,7,8,9,10,11,12,13,14,15);
}

// ---------- gfx1250 async global->LDS staging (ASYNCcnt path) ----------
#if __has_builtin(__builtin_amdgcn_global_load_async_to_lds_b128)
#define HAVE_ASYNC_LDS 1
__device__ __forceinline__ void async_cp16(const void* g, void* l) {
    __builtin_amdgcn_global_load_async_to_lds_b128(
        (__attribute__((address_space(1))) i32x4*)g,
        (__attribute__((address_space(3))) i32x4*)l, 0, 0);
}
#else
#define HAVE_ASYNC_LDS 0
#endif

template <short N>
__device__ __forceinline__ void wait_async() {
#if __has_builtin(__builtin_amdgcn_s_wait_asynccnt)
    __builtin_amdgcn_s_wait_asynccnt(N);
#else
    asm volatile("s_wait_asynccnt %0" :: "i"(N) : "memory");
#endif
}

// ---------- prep: x (fp32) -> bf16, row-major unchanged ----------
__global__ __launch_bounds__(256)
void convert_x(const float* __restrict__ x, __bf16* __restrict__ xb)
{
    const size_t idx = ((size_t)blockIdx.x * 256 + threadIdx.x) * 8;
    f32x4 a = *(const f32x4*)(x + idx);
    f32x4 b = *(const f32x4*)(x + idx + 4);
    bf16x8 v;
    v[0]=(__bf16)a[0]; v[1]=(__bf16)a[1]; v[2]=(__bf16)a[2]; v[3]=(__bf16)a[3];
    v[4]=(__bf16)b[0]; v[5]=(__bf16)b[1]; v[6]=(__bf16)b[2]; v[7]=(__bf16)b[3];
    *(bf16x8*)(xb + idx) = v;
}

// ---------- prep: W [K][N] fp32 -> Wt [N][K] bf16 (4 weights, z-indexed) ----------
__global__ __launch_bounds__(256)
void transpose_w(const float* __restrict__ W0, const float* __restrict__ W1,
                 const float* __restrict__ W2, const float* __restrict__ W3,
                 __bf16* __restrict__ T)
{
    __shared__ float t[32][33];
    const int tx = threadIdx.x & 31, ty = threadIdx.x >> 5;   // 32 x 8
    const int c0 = blockIdx.x * 32, r0 = blockIdx.y * 32, z = blockIdx.z;
    const float* W = (z == 0) ? W0 : (z == 1) ? W1 : (z == 2) ? W2 : W3;
    __bf16* Wt = T + (size_t)z * EMB * EMB;
#pragma unroll
    for (int j = 0; j < 4; ++j)
        t[ty + 8*j][tx] = W[(size_t)(r0 + ty + 8*j) * EMB + c0 + tx];
    __syncthreads();
#pragma unroll
    for (int j = 0; j < 4; ++j)
        Wt[(size_t)(c0 + ty + 8*j) * EMB + r0 + tx] = (__bf16)t[tx][ty + 8*j];
}

// C[M=8192][N=1024] = A[M][1024] * Bt^T + bias   (A, Bt both bf16; Bt is [N][K])
// Double-buffered async-LDS pipeline: tile k+1 DMA overlaps tile k WMMA.
// mode 0/1: out bf16 [b*8+h][t][d]   (Q, K layouts)
// mode 2  : out bf16 [b*8+h][d][t]   (V transposed)
// mode 3  : out f32  [row][col]      (final projection -> d_out)
__global__ __launch_bounds__(256)
void gemm_bf16(const __bf16* __restrict__ A, const __bf16* __restrict__ Bt,
               const float* __restrict__ bias,
               __bf16* __restrict__ outb, float* __restrict__ outf, int mode)
{
    constexpr int BM = 128, BN = 128, BK = 64, LDT = BK + 8;
    constexpr int NT = EMB / BK;                    // 16 k-tiles
    __shared__ __bf16 As[2][BM * LDT];
    __shared__ __bf16 Bs[2][BM * LDT];              // stored [n][k]

    const int tid  = threadIdx.x;
    const int lane = tid & 31;
    const int wave = tid >> 5;
    const int wm   = wave >> 2;       // 0..1  (64 rows each)
    const int wn   = wave & 3;        // 0..3  (32 cols each)
    const int l15  = lane & 15;
    const int lhi  = lane >> 4;       // 0 or 1
    const int rowBase = blockIdx.x * BM;
    const int colBase = blockIdx.y * BN;

    // stage one 128x64 A tile + 128x64 B tile into LDS slot (8 ops/thread)
    auto stage = [&](int slot, int kb) {
#pragma unroll
        for (int p = 0; p < 4; ++p) {
            int idx = p * 256 + tid;          // 1024 chunks of 16B per tile
            int r   = idx >> 3;               // 0..127
            int ch  = (idx & 7) * 8;          // 0..56
#if HAVE_ASYNC_LDS
            async_cp16(A  + (size_t)(rowBase + r) * EMB + kb + ch, &As[slot][r * LDT + ch]);
            async_cp16(Bt + (size_t)(colBase + r) * EMB + kb + ch, &Bs[slot][r * LDT + ch]);
#else
            *(bf16x8*)(&As[slot][r * LDT + ch]) = *(const bf16x8*)(A  + (size_t)(rowBase + r) * EMB + kb + ch);
            *(bf16x8*)(&Bs[slot][r * LDT + ch]) = *(const bf16x8*)(Bt + (size_t)(colBase + r) * EMB + kb + ch);
#endif
        }
    };

    f32x8 acc[4][2] = {};

    stage(0, 0);                                    // prologue: tile 0 in flight
    for (int t = 0; t < NT; ++t) {
        const int cur = t & 1;
        if (t + 1 < NT) {
            // issue DMA for next tile into the other slot (its readers finished
            // at the barrier that ended iteration t-1), then wait only for the
            // oldest 8 async ops (= current tile; ASYNCcnt completes in order).
            stage(1 - cur, (t + 1) * BK);
#if HAVE_ASYNC_LDS
            wait_async<8>();
#endif
        } else {
#if HAVE_ASYNC_LDS
            wait_async<0>();
#endif
        }
        __syncthreads();

        const __bf16* as = As[cur];
        const __bf16* bs = Bs[cur];
#pragma unroll
        for (int kk = 0; kk < 2; ++kk) {
            const int k0 = kk * 32;
            bf16x16 aF[4];
#pragma unroll
            for (int m = 0; m < 4; ++m) {
                const __bf16* base = &as[(wm * 64 + m * 16 + l15) * LDT + k0 + lhi * 16];
                aF[m] = cat8(*(const bf16x8*)base, *(const bf16x8*)(base + 8));
            }
            bf16x16 bF[2];
#pragma unroll
            for (int n = 0; n < 2; ++n)
                bF[n] = *(const bf16x16*)(&bs[(wn * 32 + n * 16 + l15) * LDT + k0 + lhi * 16]);
#pragma unroll
            for (int m = 0; m < 4; ++m)
#pragma unroll
                for (int n = 0; n < 2; ++n)
                    acc[m][n] = wmma_bf16(aF[m], bF[n], acc[m][n]);
        }
        __syncthreads();
    }

    // ---- epilogue: C layout = lane(l15)=col, vgpr i -> row lhi*8+i ----
#pragma unroll
    for (int m = 0; m < 4; ++m) {
        const int r0 = rowBase + wm * 64 + m * 16 + lhi * 8;
#pragma unroll
        for (int n = 0; n < 2; ++n) {
            const int c  = colBase + wn * 32 + n * 16 + l15;
            const float bv = bias[c];
            if (mode == 2) {           // V transposed: [bh][d][t], 8 consecutive t
                const int b = r0 >> 10, t = r0 & 1023;
                const int h = c >> 7,  d = c & 127;
                bf16x8 v;
#pragma unroll
                for (int i = 0; i < 8; ++i) v[i] = (__bf16)(acc[m][n][i] + bv);
                *(bf16x8*)(outb + ((size_t)((b * HEADS + h) * HD + d)) * SEQ + t) = v;
            } else if (mode == 3) {    // fp32 output
#pragma unroll
                for (int i = 0; i < 8; ++i)
                    outf[(size_t)(r0 + i) * EMB + c] = acc[m][n][i] + bv;
            } else {                   // Q/K: [bh][t][d]
                const int h = c >> 7, d = c & 127;
#pragma unroll
                for (int i = 0; i < 8; ++i) {
                    const int r = r0 + i;
                    const int b = r >> 10, t = r & 1023;
                    outb[((size_t)((b * HEADS + h) * SEQ + t)) * HD + d] = (__bf16)(acc[m][n][i] + bv);
                }
            }
        }
    }
}

// Per wave: 16 queries x full key range; unnormalized exp-accumulation,
// single normalization (1 / (rowsum * sqrt(emb))) at the end.
__global__ __launch_bounds__(256)
void attention_kernel(const __bf16* __restrict__ Q,   // [bh][t][128]
                      const __bf16* __restrict__ K,   // [bh][t][128]
                      const __bf16* __restrict__ Vt,  // [bh][128][t]
                      __bf16* __restrict__ ctx)       // [b][t][1024]
{
    __shared__ __bf16 P[8][16][40];   // per-wave 16x32 bf16 prob tile (+pad)
    const int tid  = threadIdx.x;
    const int lane = tid & 31, wave = tid >> 5;
    const int l15  = lane & 15, lhi = lane >> 4;
    const int h = blockIdx.y, b = blockIdx.z;
    const int bh = b * HEADS + h;
    const int q0 = blockIdx.x * 128 + wave * 16;

    const __bf16* Qp = Q  + (size_t)bh * SEQ * HD;
    const __bf16* Kp = K  + (size_t)bh * SEQ * HD;
    const __bf16* Vp = Vt + (size_t)bh * HD * SEQ;

    bf16x16 qF[4];                     // Q row = q0 + l15, K-dim 128 = 4 steps
#pragma unroll
    for (int kk = 0; kk < 4; ++kk) {
        const __bf16* base = Qp + (size_t)(q0 + l15) * HD + kk * 32 + lhi * 16;
        qF[kk] = cat8(*(const bf16x8*)base, *(const bf16x8*)(base + 8));
    }

    f32x8 o[8] = {};
    float rsum[8] = {0.f,0.f,0.f,0.f,0.f,0.f,0.f,0.f};

    for (int kt = 0; kt < SEQ / 32; ++kt) {
        f32x8 s0 = {}, s1 = {};
        const int key0 = kt * 32 + l15;
#pragma unroll
        for (int kk = 0; kk < 4; ++kk) {
            bf16x16 b0 = *(const bf16x16*)(Kp + (size_t)key0        * HD + kk * 32 + lhi * 16);
            bf16x16 b1 = *(const bf16x16*)(Kp + (size_t)(key0 + 16) * HD + kk * 32 + lhi * 16);
            s0 = wmma_bf16(qF[kk], b0, s0);
            s1 = wmma_bf16(qF[kk], b1, s1);
        }
        // exp, rowsum, and C-layout -> LDS (row = lhi*8+i, col = l15 / 16+l15)
#pragma unroll
        for (int i = 0; i < 8; ++i) {
            const float e0 = __expf(s0[i]);
            const float e1 = __expf(s1[i]);
            rsum[i] += e0 + e1;
            P[wave][lhi * 8 + i][l15]      = (__bf16)e0;
            P[wave][lhi * 8 + i][16 + l15] = (__bf16)e1;
        }
        // reload probs in A-matrix layout (same-wave LDS ops are in order)
        const __bf16* pb = &P[wave][l15][lhi * 16];
        bf16x16 pF = cat8(*(const bf16x8*)pb, *(const bf16x8*)(pb + 8));
        // O(16x128) += P(16x32) * V(32x128): Vt column-major over keys
#pragma unroll
        for (int n = 0; n < 8; ++n) {
            bf16x16 bv = *(const bf16x16*)(Vp + (size_t)(n * 16 + l15) * SEQ + kt * 32 + lhi * 16);
            o[n] = wmma_bf16(pF, bv, o[n]);
        }
    }

    // finish rowsums across the 16 lanes holding each row group
#pragma unroll
    for (int i = 0; i < 8; ++i) {
        float v = rsum[i];
        v += __shfl_xor(v, 1);
        v += __shfl_xor(v, 2);
        v += __shfl_xor(v, 4);
        v += __shfl_xor(v, 8);
        rsum[i] = 1.0f / (v * 32.0f);   // softmax norm * 1/sqrt(emb)
    }
#pragma unroll
    for (int n = 0; n < 8; ++n) {
        const int d = h * HD + n * 16 + l15;
#pragma unroll
        for (int i = 0; i < 8; ++i) {
            const int q = q0 + lhi * 8 + i;
            ctx[((size_t)b * SEQ + q) * EMB + d] = (__bf16)(o[n][i] * rsum[i]);
        }
    }
}

extern "C" void kernel_launch(void* const* d_in, const int* in_sizes, int n_in,
                              void* d_out, int out_size, void* d_ws, size_t ws_size,
                              hipStream_t stream)
{
    const float* x  = (const float*)d_in[0];
    const float* Wq = (const float*)d_in[1];
    const float* bq = (const float*)d_in[2];
    const float* Wk = (const float*)d_in[3];
    const float* bk = (const float*)d_in[4];
    const float* Wv = (const float*)d_in[5];
    const float* bv = (const float*)d_in[6];
    const float* Wp = (const float*)d_in[7];
    const float* bp = (const float*)d_in[8];
    float* out = (float*)d_out;

    char* ws = (char*)d_ws;
    const size_t sz = (size_t)BATCH * HEADS * SEQ * HD * sizeof(__bf16); // 16 MB
    __bf16* Qb  = (__bf16*)(ws);
    __bf16* Kb  = (__bf16*)(ws + sz);
    __bf16* Vtb = (__bf16*)(ws + 2 * sz);
    __bf16* ctx = (__bf16*)(ws + 3 * sz);
    __bf16* xb  = (__bf16*)(ws + 4 * sz);            // 16 MB
    __bf16* Wt  = (__bf16*)(ws + 5 * sz);            // 4 x 2 MB
    const size_t wsz = (size_t)EMB * EMB;

    dim3 blk(256);
    convert_x<<<dim3(4096), blk, 0, stream>>>(x, xb);
    transpose_w<<<dim3(32, 32, 4), blk, 0, stream>>>(Wq, Wk, Wv, Wp, Wt);

    dim3 g(64, 8);                                   // 8192/128 x 1024/128
    gemm_bf16<<<g, blk, 0, stream>>>(xb,  Wt + 0*wsz, bq, Qb,  nullptr, 0);
    gemm_bf16<<<g, blk, 0, stream>>>(xb,  Wt + 1*wsz, bk, Kb,  nullptr, 1);
    gemm_bf16<<<g, blk, 0, stream>>>(xb,  Wt + 2*wsz, bv, Vtb, nullptr, 2);
    attention_kernel<<<dim3(8, HEADS, BATCH), blk, 0, stream>>>(Qb, Kb, Vtb, ctx);
    gemm_bf16<<<g, blk, 0, stream>>>(ctx, Wt + 3*wsz, bp, nullptr, out, 3);
}
